// CKA_47072841564786
// MI455X (gfx1250) — matrix operand order repaired and checked by reading, not compiled, not run on hardware
//
#include <hip/hip_runtime.h>
#include <hip/hip_bf16.h>
#include <math.h>

// ---------------------------------------------------------------------------
// Linear CKA via feature-space GEMM:
//   hsic(a,b) = || Za_c^T Zb_c ||_F^2 ,  Za_c^T Zb_c = A^T B - s_a s_b^T / n
// Z = [X|Y] (8192 x 3072), G = Z^T Z computed with bf16x(hi,lo) split WMMA
// (3 products: hh + hl + lh), fp32 accumulation. Upper-triangular tiles only.
// Software-pipelined K loop with double-buffered LDS (one barrier per iter).
// ---------------------------------------------------------------------------

typedef __attribute__((ext_vector_type(16))) __bf16 v16bf;
typedef __attribute__((ext_vector_type(8)))  __bf16 v8bf;
typedef __attribute__((ext_vector_type(4)))  __bf16 v4bf;
typedef __attribute__((ext_vector_type(8)))  float  v8f;

#define N_ROWS   8192
#define DX       2048
#define DY       1024
#define DZ       3072
#define TILE     128
#define NT       24            // DZ / TILE ; X tiles: 0..15, Y tiles: 16..23
#define NTILES   300           // NT*(NT+1)/2 upper-triangular tiles
#define KSTEP    32
#define LDP      40            // padded LDS row (bf16 elems) = 80B, 16B-aligned
// ws layout (floats): [0..3071] column sums of Z ; [3072]=xx [3073]=xy [3074]=yy
#define WS_XX    3072
#define WS_XY    3073
#define WS_YY    3074
#define WS_N     3075

__device__ __forceinline__ v8f wmma_bf16(v16bf a, v16bf b, v8f c) {
  // D = A x B + C, 16x16x32 bf16 -> f32
  return __builtin_amdgcn_wmma_f32_16x16x32_bf16(false, a, false, b,
                                                 (short)0, c, false, false);
}

__device__ __forceinline__ v16bf frag_join(const __bf16* p0, const __bf16* p1) {
  v8bf c0 = *(const v8bf*)p0;   // ds_load_b128
  v8bf c1 = *(const v8bf*)p1;   // ds_load_b128
  return __builtin_shufflevector(c0, c1, 0,1,2,3,4,5,6,7,8,9,10,11,12,13,14,15);
}

// ---------------------------------------------------------------------------
__global__ void __launch_bounds__(256)
cka_init_kernel(float* __restrict__ ws) {
  for (int i = threadIdx.x; i < WS_N; i += 256) ws[i] = 0.0f;
}

// column sums of Z = [X|Y]; grid (12, 8), each y-block covers 1024 rows
__global__ void __launch_bounds__(256)
cka_colsum_kernel(const float* __restrict__ x, const float* __restrict__ y,
                  float* __restrict__ ws) {
  int c  = blockIdx.x * 256 + threadIdx.x;      // 0..3071
  int r0 = blockIdx.y * 1024;
  const float* base; int ld, col;
  if (c < DX) { base = x; ld = DX; col = c; }
  else        { base = y; ld = DY; col = c - DX; }
  float s = 0.0f;
  for (int r = r0; r < r0 + 1024; ++r)
    s += base[(size_t)r * ld + col];
  atomicAdd(&ws[c], s);
}

// ---------------------------------------------------------------------------
__global__ void __launch_bounds__(256)
cka_gemm_kernel(const float* __restrict__ x, const float* __restrict__ y,
                float* __restrict__ ws) {
  // double-buffered panels, layout [feature m][k], 80B padded rows
  __shared__ __bf16 Ahi[2][TILE][LDP], Alo[2][TILE][LDP];
  __shared__ __bf16 Bhi[2][TILE][LDP], Blo[2][TILE][LDP];
  __shared__ float  sA[TILE], sB[TILE];
  __shared__ float  red[256];

  // upper-triangular tile enumeration: blockIdx.x -> (ti, tj), tj >= ti
  int l = blockIdx.x, ti = 0;
  while (l >= NT - ti) { l -= NT - ti; ++ti; }
  const int tj = ti + l;

  const int tid = threadIdx.x;

  // column-sum slices for the centering correction
  if (tid < TILE) sA[tid]        = ws[ti * TILE + tid];
  else            sB[tid - TILE] = ws[tj * TILE + (tid - TILE)];

  // panel bases (each 128-wide tile lies wholly inside X or Y)
  const float* baseA; int ldA, colA;
  { int cb = ti * TILE;
    if (cb < DX) { baseA = x; ldA = DX; colA = cb; }
    else         { baseA = y; ldA = DY; colA = cb - DX; } }
  const float* baseB; int ldB, colB;
  { int cb = tj * TILE;
    if (cb < DX) { baseB = x; ldB = DX; colB = cb; }
    else         { baseB = y; ldB = DY; colB = cb - DX; } }

  const int lane  = tid & 31;
  const int lm    = lane & 15;
  const int lhalf = lane >> 4;       // wave32: half-wave select in WMMA layouts
  const int wid   = tid >> 5;        // 8 waves
  const int waveM = wid >> 2;        // 0..1  -> 64-row slab
  const int waveN = wid & 3;         // 0..3  -> 32-col slab

  v8f acc[4][2] = {};                // 4 (M) x 2 (N) 16x16 f32 accumulators

  const int stM = tid & 127;         // staging: feature index
  const int stK = (tid >> 7) << 2;   // staging: k0 base (0 or 4)
  const float* pA = baseA + colA + stM;
  const float* pB = baseB + colB + stM;

  float fa[16], fb[16];              // in-flight panel (software pipeline)

  // issue the global loads for a 32x128 A panel + B panel at row kbase
  auto load_panel = [&](int kbase) {
#pragma unroll
    for (int it = 0; it < 4; ++it) {
      const int k0 = stK + it * 8;
#pragma unroll
      for (int j = 0; j < 4; ++j) {
        fa[it * 4 + j] = pA[(size_t)(kbase + k0 + j) * ldA];
        fb[it * 4 + j] = pB[(size_t)(kbase + k0 + j) * ldB];
      }
    }
  };

  // fp32 -> bf16 hi/lo split, store into LDS buffer `buf`
  auto store_panel = [&](int buf) {
#pragma unroll
    for (int it = 0; it < 4; ++it) {
      const int k0 = stK + it * 8;
      v4bf ah, al, bh, bl;
#pragma unroll
      for (int j = 0; j < 4; ++j) {
        float f = fa[it * 4 + j];
        __bf16 h = (__bf16)f;
        ah[j] = h;  al[j] = (__bf16)(f - (float)h);
        float g = fb[it * 4 + j];
        __bf16 hg = (__bf16)g;
        bh[j] = hg; bl[j] = (__bf16)(g - (float)hg);
      }
      *(v4bf*)&Ahi[buf][stM][k0] = ah;
      *(v4bf*)&Alo[buf][stM][k0] = al;
      *(v4bf*)&Bhi[buf][stM][k0] = bh;
      *(v4bf*)&Blo[buf][stM][k0] = bl;
    }
  };

  // fragment loads + 24 WMMAs from LDS buffer `buf`
  auto compute = [&](int buf) {
    // B fragments (K x 16): lanes 0-15 hold K=0..15, lanes 16-31 K=16..31
    v16bf bhf[2], blf[2];
#pragma unroll
    for (int ns = 0; ns < 2; ++ns) {
      const __bf16* rh = &Bhi[buf][waveN * 32 + ns * 16 + lm][0];
      const __bf16* rl = &Blo[buf][waveN * 32 + ns * 16 + lm][0];
      bhf[ns] = frag_join(rh + lhalf * 16, rh + lhalf * 16 + 8);
      blf[ns] = frag_join(rl + lhalf * 16, rl + lhalf * 16 + 8);
    }
    // A fragments (16 x K): lanes 0-15 K=0..15, lanes 16-31 K=16..31
#pragma unroll
    for (int ms = 0; ms < 4; ++ms) {
      const __bf16* rh = &Ahi[buf][waveM * 64 + ms * 16 + lm][0];
      const __bf16* rl = &Alo[buf][waveM * 64 + ms * 16 + lm][0];
      v16bf ah = frag_join(rh + lhalf * 16, rh + lhalf * 16 + 8);
      v16bf al = frag_join(rl + lhalf * 16, rl + lhalf * 16 + 8);
#pragma unroll
      for (int ns = 0; ns < 2; ++ns) {
        acc[ms][ns] = wmma_bf16(ah, bhf[ns], acc[ms][ns]);   // hi*hi
        acc[ms][ns] = wmma_bf16(ah, blf[ns], acc[ms][ns]);   // hi*lo
        acc[ms][ns] = wmma_bf16(al, bhf[ns], acc[ms][ns]);   // lo*hi
      }
    }
  };

  // ---- software-pipelined main loop: global loads overlap WMMA compute ----
  load_panel(0);
  store_panel(0);
  __syncthreads();

  int cur = 0;
  for (int kb = 0; kb < N_ROWS; kb += KSTEP) {
    const bool not_last = (kb + KSTEP < N_ROWS);   // uniform branch
    if (not_last) load_panel(kb + KSTEP);          // issue loads, no wait yet
    compute(cur);                                  // WMMAs on current buffer
    if (not_last) store_panel(cur ^ 1);            // waits loads, fills other
    __syncthreads();
    cur ^= 1;
  }

  // ---- epilogue: centering correction + squared Frobenius partial
  const float inv_n = 1.0f / (float)N_ROWS;
  float sq = 0.0f;
#pragma unroll
  for (int ms = 0; ms < 4; ++ms)
#pragma unroll
    for (int ns = 0; ns < 2; ++ns)
#pragma unroll
      for (int p = 0; p < 8; ++p) {
        // C layout: VGPR p -> M = p (lanes 0-15) / p+8 (lanes 16-31); N = lane%16
        int ml = waveM * 64 + ms * 16 + p + (lhalf ? 8 : 0);
        int nl = waveN * 32 + ns * 16 + lm;
        float g = acc[ms][ns][p] - sA[ml] * sB[nl] * inv_n;
        sq += g * g;
      }
  red[tid] = sq;
  __syncthreads();
  for (int s = 128; s > 0; s >>= 1) {
    if (tid < s) red[tid] += red[tid + s];
    __syncthreads();
  }
  if (tid == 0) {
    int bucket; float w;
    if (tj < 16)      { bucket = WS_XX; w = (ti == tj) ? 1.0f : 2.0f; }
    else if (ti < 16) { bucket = WS_XY; w = 1.0f; }   // X-Y block appears once
    else              { bucket = WS_YY; w = (ti == tj) ? 1.0f : 2.0f; }
    atomicAdd(&ws[bucket], w * red[0]);
  }
}

// ---------------------------------------------------------------------------
__global__ void cka_final_kernel(const float* __restrict__ ws,
                                 float* __restrict__ out) {
  float xx = ws[WS_XX], xy = ws[WS_XY], yy = ws[WS_YY];
  out[0] = xy / (sqrtf(xx * yy) + 1e-8f);
}

// ---------------------------------------------------------------------------
extern "C" void kernel_launch(void* const* d_in, const int* in_sizes, int n_in,
                              void* d_out, int out_size, void* d_ws, size_t ws_size,
                              hipStream_t stream) {
  const float* x = (const float*)d_in[0];   // 8192 x 2048
  const float* y = (const float*)d_in[1];   // 8192 x 1024
  float* ws  = (float*)d_ws;                // needs >= WS_N floats (~12.3 KB)
  float* out = (float*)d_out;

  cka_init_kernel<<<1, 256, 0, stream>>>(ws);
  cka_colsum_kernel<<<dim3(DZ / 256, 8), 256, 0, stream>>>(x, y, ws);
  cka_gemm_kernel<<<NTILES, 256, 0, stream>>>(x, y, ws);
  cka_final_kernel<<<1, 1, 0, stream>>>(ws, out);
}
